// EHH_Layer_9388798509377
// MI455X (gfx1250) — matrix operand order repaired and checked by reading, not compiled, not run on hardware
//
#include <hip/hip_runtime.h>
#include <hip/hip_bf16.h>

typedef __attribute__((ext_vector_type(16))) _Float16 v16h;
typedef __attribute__((ext_vector_type(8)))  float    v8f;

#if __has_builtin(__builtin_amdgcn_sched_barrier)
#define SCHED_FENCE() __builtin_amdgcn_sched_barrier(0)
#else
#define SCHED_FENCE()
#endif

#define B_ROWS 8192
#define N_COLS 512
#define QQ     6
#define MM     1024
#define OUTN   256
#define KDIM   4096   // N*Q + M
#define KT_CNT 128    // KDIM / 32
#define MT_CNT 512    // B_ROWS / 16

// ---------------------------------------------------------------------------
// Kernel 1: partial column sums (no atomics -> deterministic).
// grid (2, 32), block 256. Block y handles 256 rows; thread x handles one col.
// ---------------------------------------------------------------------------
__global__ __launch_bounds__(256) void stats_partial(const float* __restrict__ x,
                                                     float* __restrict__ psum,
                                                     float* __restrict__ psumsq) {
    int col = blockIdx.x * 256 + threadIdx.x;        // 0..511
    int r0  = blockIdx.y * 256;
    const float* p = x + (size_t)r0 * N_COLS + col;
    float s = 0.f, ss = 0.f;
    for (int i = 0; i < 256; ++i) {
        float v = p[(size_t)i * N_COLS];
        s += v;
        ss += v * v;
    }
    psum  [blockIdx.y * N_COLS + col] = s;
    psumsq[blockIdx.y * N_COLS + col] = ss;
}

// ---------------------------------------------------------------------------
// Kernel 2: finalize stats -> per-column scale and 6 offsets.
//   normed      = scale*x + (beta - scale*mean),          scale = gamma*rstd
//   branch q>=1 = scale*x + off0 + beta - gamma*(c*var+mean)
// ---------------------------------------------------------------------------
__global__ __launch_bounds__(256) void stats_final(const float* __restrict__ psum,
                                                   const float* __restrict__ psumsq,
                                                   const float* __restrict__ beta,
                                                   const float* __restrict__ gamma,
                                                   float* __restrict__ scale,
                                                   float* __restrict__ offs) {
    int col = blockIdx.x * 256 + threadIdx.x;        // 0..511
    float s = 0.f, ss = 0.f;
    for (int c = 0; c < 32; ++c) {
        s  += psum  [c * N_COLS + col];
        ss += psumsq[c * N_COLS + col];
    }
    const float inv  = 1.0f / (float)B_ROWS;
    float mean = s * inv;
    float var  = ss * inv - mean * mean;             // ddof=0, matches jnp.var
    float rstd = rsqrtf(var + 0.001f);
    float g  = gamma[col];
    float bt = beta[col];
    float sc = g * rstd;
    scale[col] = sc;
    float o0 = bt - sc * mean;
    offs[col * QQ + 0] = o0;
    const float C[5] = {-3.0f, -0.834f, -0.248f, 0.248f, 0.834f};
#pragma unroll
    for (int j = 0; j < 5; ++j)
        offs[col * QQ + 1 + j] = o0 + bt - g * (C[j] * var + mean);
}

// ---------------------------------------------------------------------------
// Kernel 3: pack w (4096x256 f32, row-major K x N) into f16 WMMA B panels.
// B layout (16-bit 32x16): lane l -> N = l%16; K = (h<8 ? h : h+8) + (l<16?0:8).
// Panel index: [kt][nt][lane][h], each panel = 32 lanes * 16 halves.
// ---------------------------------------------------------------------------
__global__ __launch_bounds__(256) void pack_w(const float* __restrict__ w,
                                              _Float16* __restrict__ wpk) {
    int id   = blockIdx.x * 256 + threadIdx.x;       // half index
    int h    = id & 15;
    int lane = (id >> 4) & 31;
    int nt   = (id >> 9) & 15;
    int kt   = id >> 13;
    int k = kt * 32 + ((lane & 16) ? 8 : 0) + (h < 8 ? h : h + 8);
    int n = nt * 16 + (lane & 15);
    wpk[id] = (_Float16)w[(size_t)k * OUTN + n];
}

// ---------------------------------------------------------------------------
// Kernel 4: build feat directly in f16 WMMA A-panel layout, fused math.
// A layout (16-bit 16x32): lane l -> M = l%16; K = (h<8 ? h : h+8) + (l<16?0:8).
// Panel index: [mtile][ktile][lane][16 halves] -> one v16h (32B) per thread.
// ---------------------------------------------------------------------------
__global__ __launch_bounds__(256) void pack_feat(const float* __restrict__ x,
                                                 const float* __restrict__ scale,
                                                 const float* __restrict__ offs,
                                                 const int*   __restrict__ ci,
                                                 v16h* __restrict__ featpk) {
    int id    = blockIdx.x * 256 + threadIdx.x;      // v16h index
    int lane  = id & 31;
    int t     = id >> 5;
    int ktile = t & (KT_CNT - 1);
    int mtile = t >> 7;
    int row   = mtile * 16 + (lane & 15);
    const float* xr = x + (size_t)row * N_COLS;
    int kbase = ktile * 32 + ((lane & 16) ? 8 : 0);
    v16h out;
#pragma unroll
    for (int h = 0; h < 16; ++h) {
        int k = kbase + (h < 8 ? h : h + 8);
        float v;
        if (k < N_COLS * QQ) {
            int n = k / QQ;
            int q = k - n * QQ;
            v = fmaxf(0.f, scale[n] * xr[n] + offs[n * QQ + q]);
        } else {
            int m = k - N_COLS * QQ;
            int n1 = ci[m * 5 + 1], q1 = ci[m * 5 + 2];
            int n2 = ci[m * 5 + 3], q2 = ci[m * 5 + 4];
            float v1 = fmaxf(0.f, scale[n1] * xr[n1] + offs[n1 * QQ + q1]);
            float v2 = fmaxf(0.f, scale[n2] * xr[n2] + offs[n2 * QQ + q2]);
            v = fminf(v1, v2);
        }
        out[h] = (_Float16)v;
    }
    featpk[id] = out;
}

// ---------------------------------------------------------------------------
// Kernel 5: GEMM via v_wmma_f32_16x16x32_f16, manual double buffering with
// sched_barrier fences pinning [loads kt+1][WMMA kt][loads kt+2][WMMA kt+1].
// 128 blocks x 256 threads (8 waves as 2M x 4N). Wave tile: 32 rows x 64 cols
// = 2 mtiles x 4 ntiles = 8 accumulators; 6 panel loads per 8 WMMAs.
// Block tile: 64 rows x 256 cols -> feat streamed exactly once.
// ---------------------------------------------------------------------------
__global__ __launch_bounds__(256) void gemm_wmma(const v16h* __restrict__ featpk,
                                                 const v16h* __restrict__ wpk,
                                                 const float* __restrict__ bias,
                                                 float* __restrict__ out) {
    int lane = threadIdx.x & 31;
    int wid  = threadIdx.x >> 5;
    int wm   = wid & 1;                              // M wave 0..1
    int wn   = wid >> 1;                             // N wave 0..3
    int mt0  = blockIdx.x * 4 + wm * 2;              // first of 2 global mtiles
    int nt0  = wn * 4;                               // first of 4 global ntiles

    const v16h* ap0 = featpk + (size_t)(mt0 + 0) * KT_CNT * 32 + lane;
    const v16h* ap1 = featpk + (size_t)(mt0 + 1) * KT_CNT * 32 + lane;
    const v16h* bp  = wpk + (size_t)nt0 * 32 + lane;

    v8f acc[2][4] = {};
    v16h a0[2], b0[4], a1[2], b1[4];

    // Preload kt = 0 into buffer 0.
    a0[0] = ap0[0];
    a0[1] = ap1[0];
#pragma unroll
    for (int nt = 0; nt < 4; ++nt) b0[nt] = bp[(size_t)nt * 32];

#pragma unroll 1
    for (int kt = 0; kt < KT_CNT; kt += 2) {
        // Issue loads for kt+1 into buffer 1 (always in range: kt+1 <= 127).
        int ktA = kt + 1;
        a1[0] = ap0[(size_t)ktA * 32];
        a1[1] = ap1[(size_t)ktA * 32];
#pragma unroll
        for (int nt = 0; nt < 4; ++nt)
            b1[nt] = bp[(size_t)(ktA * 16 + nt) * 32];

        SCHED_FENCE();   // loads + their address math stay above the WMMAs

        // Compute with buffer 0 (kt) -- waits only on *last* iteration's loads.
#pragma unroll
        for (int mi = 0; mi < 2; ++mi)
#pragma unroll
            for (int nt = 0; nt < 4; ++nt)
                acc[mi][nt] = __builtin_amdgcn_wmma_f32_16x16x32_f16(
                    false, (mi ? a0[1] : a0[0]), false, b0[nt],
                    (short)0, acc[mi][nt], false, false);

        SCHED_FENCE();

        // Issue loads for kt+2 into buffer 0 (clamped at tail; redundant load).
        int ktB = (kt + 2 < KT_CNT) ? kt + 2 : KT_CNT - 1;
        a0[0] = ap0[(size_t)ktB * 32];
        a0[1] = ap1[(size_t)ktB * 32];
#pragma unroll
        for (int nt = 0; nt < 4; ++nt)
            b0[nt] = bp[(size_t)(ktB * 16 + nt) * 32];

        SCHED_FENCE();

        // Compute with buffer 1 (kt+1).
#pragma unroll
        for (int mi = 0; mi < 2; ++mi)
#pragma unroll
            for (int nt = 0; nt < 4; ++nt)
                acc[mi][nt] = __builtin_amdgcn_wmma_f32_16x16x32_f16(
                    false, (mi ? a1[1] : a1[0]), false, b1[nt],
                    (short)0, acc[mi][nt], false, false);

        SCHED_FENCE();
    }

    float bs = bias[0];
    // C/D layout: VGPR r, lanes 0-15 -> (M=r, N=lane), lanes 16-31 -> (M=r+8).
    int colb = nt0 * 16 + (lane & 15);
#pragma unroll
    for (int mi = 0; mi < 2; ++mi) {
        int rowb = (mt0 + mi) * 16 + ((lane & 16) ? 8 : 0);
#pragma unroll
        for (int nt = 0; nt < 4; ++nt) {
#pragma unroll
            for (int r = 0; r < 8; ++r) {
                out[(size_t)(rowb + r) * OUTN + colb + nt * 16] = acc[mi][nt][r] + bs;
            }
        }
    }
}

// ---------------------------------------------------------------------------
// Launch. Inputs: x, beta, gamma, w, biases, chosen_index.
// Workspace layout (bytes):
//   psum   [32*512] f32 @ 0
//   psumsq [32*512] f32 @ 65536
//   scale  [512]    f32 @ 131072
//   offs   [512*6]  f32 @ 133120
//   wpk    [4096*256] f16 @ 145408   (32B aligned)
//   featpk [8192*4096] f16 @ 2242560 (32B aligned), 64 MiB
// ---------------------------------------------------------------------------
extern "C" void kernel_launch(void* const* d_in, const int* in_sizes, int n_in,
                              void* d_out, int out_size, void* d_ws, size_t ws_size,
                              hipStream_t stream) {
    const float* x      = (const float*)d_in[0];
    const float* beta   = (const float*)d_in[1];
    const float* gamma  = (const float*)d_in[2];
    const float* w      = (const float*)d_in[3];
    const float* biases = (const float*)d_in[4];
    const int*   ci     = (const int*)d_in[5];

    char* ws = (char*)d_ws;
    float*    psum   = (float*)ws;
    float*    psumsq = psum + 32 * N_COLS;
    float*    scale  = psumsq + 32 * N_COLS;
    float*    offs   = scale + N_COLS;
    _Float16* wpk    = (_Float16*)(offs + N_COLS * QQ);
    v16h*     featpk = (v16h*)((char*)wpk + (size_t)KDIM * OUTN * sizeof(_Float16));

    stats_partial<<<dim3(2, 32), 256, 0, stream>>>(x, psum, psumsq);
    stats_final<<<2, 256, 0, stream>>>(psum, psumsq, beta, gamma, scale, offs);
    pack_w<<<(KDIM * OUTN) / 256, 256, 0, stream>>>(w, wpk);
    pack_feat<<<(MT_CNT * KT_CNT * 32) / 256, 256, 0, stream>>>(x, scale, offs, ci, featpk);
    gemm_wmma<<<MT_CNT / 4, 256, 0, stream>>>(featpk, (const v16h*)wpk, biases, (float*)d_out);
}
